// MLADecoderLayer_52948356825287
// MI455X (gfx1250) — compile-verified, matrix operand order checked by reading, not run on hardware
//
#include <hip/hip_runtime.h>
#include <hip/hip_bf16.h>

#define SEQ 1024
#define BATCH 2
#define NHEAD 16
#define DMODEL 1024
#define QLORA 20
#define KVLORA 20
#define ROPE_D 32
#define NOPE_D 32
#define VDIM 64
#define NEXP 4
#define HID 1024
#define TOK (SEQ*BATCH)

typedef __attribute__((ext_vector_type(16))) __bf16 v16bf;
typedef __attribute__((ext_vector_type(8)))  float v8f;
typedef __attribute__((ext_vector_type(8)))  unsigned int v8u;
typedef __attribute__((ext_vector_type(4)))  int v4i;

#if __has_builtin(__builtin_amdgcn_global_load_async_to_lds_b128) && __has_builtin(__builtin_amdgcn_s_wait_asynccnt)
#define HAS_ASYNC_LDS 1
typedef __attribute__((address_space(1))) v4i* gptr128;   // global (AS1) 128-bit chunk
typedef __attribute__((address_space(3))) v4i* lptr128;   // LDS (AS3) 128-bit chunk
#else
#define HAS_ASYNC_LDS 0
#endif

__device__ __forceinline__ unsigned short f2bf(float f) {
    unsigned int u = __builtin_bit_cast(unsigned int, f);
    unsigned int r = u + 0x7FFFu + ((u >> 16) & 1u);
    return (unsigned short)(r >> 16);
}
__device__ __forceinline__ float bf2f(unsigned short h) {
    unsigned int u = ((unsigned int)h) << 16;
    return __builtin_bit_cast(float, u);
}

// Load one 16x32 bf16 A-style fragment (also valid as 32x16 B-fragment when
// rows are the N-columns of a transposed B tile). Row stride in elements must
// make rows 16B aligned (stride 40 in LDS, 64/1024 in global).
__device__ __forceinline__ v16bf load_frag(const unsigned short* base, int stride) {
    int lane = threadIdx.x & 31;
    const unsigned short* r = base + (size_t)(lane & 15) * stride + ((lane >> 4) << 3);
    uint4 x = *(const uint4*)r;        // K pairs 0..7  (for this lane half)
    uint4 y = *(const uint4*)(r + 16); // K pairs 16..23
    v8u u; u[0]=x.x; u[1]=x.y; u[2]=x.z; u[3]=x.w; u[4]=y.x; u[5]=y.y; u[6]=y.z; u[7]=y.w;
    return __builtin_bit_cast(v16bf, u);
}

__device__ __forceinline__ v8f wmma_bf16(v16bf a, v16bf b, v8f c) {
    return __builtin_amdgcn_wmma_f32_16x16x32_bf16(false, a, false, b, (short)0, c, false, false);
}

// ---------------- f32 -> bf16 conversion ----------------
__global__ void k_f2bf(const float* __restrict__ s, unsigned short* __restrict__ d, int n) {
    int i = blockIdx.x * 256 + threadIdx.x;
    if (i < n) d[i] = f2bf(s[i]);
}

// ---------------- LoRA down-proj + RMS norm ----------------
__global__ __launch_bounds__(256) void k_lora(const float* __restrict__ dec,
        const float* __restrict__ wq_a, const float* __restrict__ qnw,
        const float* __restrict__ wkv_a, const float* __restrict__ kvnw,
        float* __restrict__ qn, float* __restrict__ ckvn, float* __restrict__ kpe) {
    int t = blockIdx.x;           // t = b*SEQ + s
    int b = t / SEQ, s = t % SEQ;
    __shared__ float xs[DMODEL];
    __shared__ float part[2][72];
    __shared__ float outs[72];
    __shared__ float rfac[2];
    const float* xp = dec + ((size_t)s * BATCH + b) * DMODEL;
    for (int i = threadIdx.x; i < DMODEL; i += 256) xs[i] = xp[i];
    __syncthreads();
    if (threadIdx.x < 144) {
        int j = threadIdx.x % 72, half = threadIdx.x / 72;
        int d0 = half * 512, d1 = d0 + 512;
        float acc = 0.f;
        if (j < QLORA) { for (int d = d0; d < d1; ++d) acc += xs[d] * wq_a[d * QLORA + j]; }
        else { int jj = j - QLORA; for (int d = d0; d < d1; ++d) acc += xs[d] * wkv_a[d * (KVLORA + ROPE_D) + jj]; }
        part[half][j] = acc;
    }
    __syncthreads();
    if (threadIdx.x < 72) outs[threadIdx.x] = part[0][threadIdx.x] + part[1][threadIdx.x];
    __syncthreads();
    if (threadIdx.x < 2) {
        float ss = 0.f; int base = threadIdx.x * QLORA;
        for (int j = 0; j < QLORA; ++j) { float v = outs[base + j]; ss += v * v; }
        rfac[threadIdx.x] = rsqrtf(ss / (float)QLORA + 1e-6f);
    }
    __syncthreads();
    if (threadIdx.x < QLORA)
        qn[(size_t)t * QLORA + threadIdx.x] = outs[threadIdx.x] * rfac[0] * qnw[threadIdx.x];
    else if (threadIdx.x < 2 * QLORA) {
        int j = threadIdx.x - QLORA;
        ckvn[(size_t)t * KVLORA + j] = outs[QLORA + j] * rfac[1] * kvnw[j];
    } else if (threadIdx.x < 2 * QLORA + ROPE_D) {
        int j = threadIdx.x - 2 * QLORA;
        kpe[(size_t)t * ROPE_D + j] = outs[2 * QLORA + j];
    }
}

// ---------------- LoRA up-proj + RoPE + assemble qf/kf/VT (bf16) ----------------
__global__ __launch_bounds__(256) void k_qkv_rope(const float* __restrict__ qn,
        const float* __restrict__ ckvn, const float* __restrict__ kpe,
        const float* __restrict__ wq_b, const float* __restrict__ wkv_b,
        unsigned short* __restrict__ qf, unsigned short* __restrict__ kf,
        unsigned short* __restrict__ vt) {
    int t = blockIdx.x; int b = t / SEQ, s = t % SEQ;
    __shared__ float q[NHEAD * 64];
    __shared__ float kvb[NHEAD * 96];
    __shared__ float qns[QLORA], cks[KVLORA], kps[ROPE_D];
    if (threadIdx.x < QLORA) qns[threadIdx.x] = qn[(size_t)t * QLORA + threadIdx.x];
    if (threadIdx.x >= 32 && threadIdx.x < 32 + KVLORA) cks[threadIdx.x - 32] = ckvn[(size_t)t * KVLORA + threadIdx.x - 32];
    if (threadIdx.x >= 64 && threadIdx.x < 64 + ROPE_D) kps[threadIdx.x - 64] = kpe[(size_t)t * ROPE_D + threadIdx.x - 64];
    __syncthreads();
    for (int o = threadIdx.x; o < NHEAD * 64; o += 256) {
        float a = 0.f;
        #pragma unroll
        for (int j = 0; j < QLORA; ++j) a += qns[j] * wq_b[j * 1024 + o];
        q[o] = a;
    }
    for (int o = threadIdx.x; o < NHEAD * 96; o += 256) {
        float a = 0.f;
        #pragma unroll
        for (int j = 0; j < KVLORA; ++j) a += cks[j] * wkv_b[j * 1536 + o];
        kvb[o] = a;
    }
    __syncthreads();
    const float LOGC = 9.210340372f / 256.0f;   // ln(10000)/256
    for (int idx = threadIdx.x; idx < NHEAD * 64; idx += 256) {
        int h = idx >> 6, j = idx & 63;
        size_t rowq = ((size_t)(b * NHEAD + h) * SEQ + s) * 64 + j;
        float qv, kv_;
        if (j < NOPE_D) { qv = q[h * 64 + j]; kv_ = kvb[h * 96 + j]; }
        else {
            int j2 = (j - NOPE_D) >> 1; int odd = j & 1;
            float fr = __expf(-(float)(h * 16 + j2) * LOGC);
            float ang = (float)s * fr, c, sn;
            __sincosf(ang, &sn, &c);
            float qx0 = q[h * 64 + NOPE_D + 2 * j2], qx1 = q[h * 64 + NOPE_D + 2 * j2 + 1];
            float kx0 = kps[2 * j2], kx1 = kps[2 * j2 + 1];
            qv  = odd ? (qx0 * sn + qx1 * c) : (qx0 * c - qx1 * sn);
            kv_ = odd ? (kx0 * sn + kx1 * c) : (kx0 * c - kx1 * sn);
        }
        qf[rowq] = f2bf(qv * 0.125f);   // fold 1/sqrt(64)
        kf[rowq] = f2bf(kv_);
        vt[((size_t)(b * NHEAD + h) * 64 + j) * SEQ + s] = f2bf(kvb[h * 96 + 32 + j]);
    }
}

// ---------------- Flash attention (bf16 WMMA, online softmax) ----------------
__global__ __launch_bounds__(128) void k_flash(const unsigned short* __restrict__ qf,
        const unsigned short* __restrict__ kf, const unsigned short* __restrict__ vt,
        unsigned short* __restrict__ xattn) {
    int bh = blockIdx.y; int b = bh >> 4, h = bh & 15;
    int wave = threadIdx.x >> 5, lane = threadIdx.x & 31;
    int qbase = blockIdx.x * 64 + wave * 16;
    const unsigned short* qp = qf + ((size_t)bh * SEQ + qbase) * 64;
    const unsigned short* kp = kf + (size_t)bh * SEQ * 64;
    const unsigned short* vp = vt + (size_t)bh * 64 * SEQ;
    v16bf qa0 = load_frag(qp, 64);
    v16bf qa1 = load_frag(qp + 32, 64);
    v8f o0 = {}, o1 = {}, o2 = {}, o3 = {};
    float m[8], l[8];
    #pragma unroll
    for (int i = 0; i < 8; ++i) { m[i] = -1e30f; l[i] = 0.f; }
    __shared__ __align__(16) unsigned short P[4][16 * 40];
    unsigned short* Pw = &P[wave][0];
    for (int kb = 0; kb < SEQ; kb += 32) {
        const unsigned short* kt = kp + (size_t)kb * 64;
        if (kb + 32 < SEQ) {   // hide L2 latency for next K/V tiles
            __builtin_prefetch(kt + 32 * 64 + (lane << 6), 0, 0);
            __builtin_prefetch(vp + kb + 32 + (size_t)(lane << 6), 0, 0);
        }
        v8f s0 = {}, s1 = {};
        { v16bf k0 = load_frag(kt, 64), k1 = load_frag(kt + 32, 64);
          s0 = wmma_bf16(qa0, k0, s0); s0 = wmma_bf16(qa1, k1, s0); }
        { v16bf k0 = load_frag(kt + 16 * 64, 64), k1 = load_frag(kt + 16 * 64 + 32, 64);
          s1 = wmma_bf16(qa0, k0, s1); s1 = wmma_bf16(qa1, k1, s1); }
        float p0[8], p1[8];
        #pragma unroll
        for (int i = 0; i < 8; ++i) {
            float mx = fmaxf(s0[i], s1[i]);
            mx = fmaxf(mx, __shfl_xor(mx, 1, 32)); mx = fmaxf(mx, __shfl_xor(mx, 2, 32));
            mx = fmaxf(mx, __shfl_xor(mx, 4, 32)); mx = fmaxf(mx, __shfl_xor(mx, 8, 32));
            float mn = fmaxf(m[i], mx);
            float alpha = __expf(m[i] - mn);
            p0[i] = __expf(s0[i] - mn); p1[i] = __expf(s1[i] - mn);
            float rs = p0[i] + p1[i];
            rs += __shfl_xor(rs, 1, 32); rs += __shfl_xor(rs, 2, 32);
            rs += __shfl_xor(rs, 4, 32); rs += __shfl_xor(rs, 8, 32);
            l[i] = l[i] * alpha + rs; m[i] = mn;
            o0[i] *= alpha; o1[i] *= alpha; o2[i] *= alpha; o3[i] *= alpha;
        }
        // transpose P (C-layout) -> A-layout via per-wave LDS scratch
        #pragma unroll
        for (int i = 0; i < 8; ++i) {
            int r = i + ((lane >> 4) << 3);
            Pw[r * 40 + (lane & 15)]      = f2bf(p0[i]);
            Pw[r * 40 + 16 + (lane & 15)] = f2bf(p1[i]);
        }
        asm volatile("s_wait_dscnt 0x0" ::: "memory");   // wave-local RAW on LDS
        v16bf pa = load_frag(Pw, 40);
        const unsigned short* vtile = vp + kb;
        o0 = wmma_bf16(pa, load_frag(vtile,            SEQ), o0);
        o1 = wmma_bf16(pa, load_frag(vtile + 16 * SEQ, SEQ), o1);
        o2 = wmma_bf16(pa, load_frag(vtile + 32 * SEQ, SEQ), o2);
        o3 = wmma_bf16(pa, load_frag(vtile + 48 * SEQ, SEQ), o3);
    }
    #pragma unroll
    for (int i = 0; i < 8; ++i) {
        float inv = 1.0f / l[i];
        int srow = qbase + i + ((lane >> 4) << 3);
        size_t base = ((size_t)srow * BATCH + b) * DMODEL + h * 64;  // (s,b) row order
        int n = lane & 15;
        xattn[base + n]      = f2bf(o0[i] * inv);
        xattn[base + 16 + n] = f2bf(o1[i] * inv);
        xattn[base + 32 + n] = f2bf(o2[i] * inv);
        xattn[base + 48 + n] = f2bf(o3[i] * inv);
    }
}

// ---------------- Generic 64x64-tile bf16 WMMA GEMM ----------------
// MODE 0: write bf16 C (DUAL -> silu(g)*u fused)   MODE 1: write f32 C
// MODE 2: scatter-add f32: out[perm[row]] += wts[perm[row]] * val
template<int MODE, bool DUAL, bool GATHER>
__global__ __launch_bounds__(128) void k_gemm(
        const unsigned short* __restrict__ A,
        const unsigned short* __restrict__ B0,
        const unsigned short* __restrict__ B1,
        void* __restrict__ Cout,
        const int* __restrict__ perm, const int* __restrict__ cnt,
        const float* __restrict__ wts,
        int M, int N, int K,
        long aExpStride, long bExpStride, long cExpStride, int permStride) {
    int z = blockIdx.z;
    int Mvalid = cnt ? cnt[z] : M;
    int mbase = blockIdx.y * 64;
    if (mbase >= Mvalid) return;
    int nbase = blockIdx.x * 64;
    const unsigned short* Ap  = A + (size_t)z * aExpStride;
    const unsigned short* Bp0 = B0 + (size_t)z * bExpStride;
    const unsigned short* Bp1 = DUAL ? (B1 + (size_t)z * bExpStride) : nullptr;
    const int* pp = perm ? (perm + (size_t)z * permStride) : nullptr;
    __shared__ __align__(16) unsigned short As[64 * 40];
    __shared__ __align__(16) unsigned short Bt[64 * 40];
    __shared__ __align__(16) unsigned short B1t[DUAL ? 64 * 40 : 16];
    int tid = threadIdx.x, lane = tid & 31, wave = tid >> 5;
    int wm = wave & 1, wn = wave >> 1;
    v8f acc[2][2] = {}, acu[2][2] = {};
    for (int kb = 0; kb < K; kb += 32) {
        __syncthreads();
#if HAS_ASYNC_LDS
        // A tile: 64 rows x 4 chunks of 16B, DMA'd global->LDS (ASYNCcnt path).
        // Out-of-range rows are clamped to a valid row; epilogue masks them.
        #pragma unroll
        for (int j = 0; j < 2; ++j) {
            int qid = tid + 128 * j;               // 0..255
            int r = qid >> 2, cc = qid & 3;        // row, 16B chunk
            int gr = mbase + r;
            int cr = (gr < Mvalid) ? gr : (Mvalid - 1);
            int arow = GATHER ? pp[cr] : cr;
            const unsigned short* gp = Ap + (size_t)arow * K + kb + 8 * cc;
            __builtin_amdgcn_global_load_async_to_lds_b128(
                (gptr128)gp, (lptr128)&As[r * 40 + 8 * cc], 0, 0);
        }
#else
        #pragma unroll
        for (int j = 0; j < 8; ++j) {              // A tile: 64 rows x 16 u32
            int qid = tid + 128 * j;
            int r = qid >> 4, c = qid & 15;
            int gr = mbase + r;
            unsigned int val = 0;
            if (gr < Mvalid) {
                int arow = GATHER ? pp[gr] : gr;
                val = *(const unsigned int*)(Ap + (size_t)arow * K + kb + 2 * c);
            }
            *(unsigned int*)&As[r * 40 + 2 * c] = val;
        }
#endif
        #pragma unroll
        for (int j = 0; j < 4; ++j) {              // B tile: 32 k x 32 n, transposed
            int qid = tid + 128 * j;
            int kk = qid >> 4, c = qid & 15;
            unsigned int v0 = *(const unsigned int*)(Bp0 + (size_t)(kb + kk) * N + nbase + 2 * c);
            Bt[(2 * c) * 40 + kk]     = (unsigned short)v0;
            Bt[(2 * c + 1) * 40 + kk] = (unsigned short)(v0 >> 16);
            if (DUAL) {
                unsigned int v1 = *(const unsigned int*)(Bp1 + (size_t)(kb + kk) * N + nbase + 2 * c);
                B1t[(2 * c) * 40 + kk]     = (unsigned short)v1;
                B1t[(2 * c + 1) * 40 + kk] = (unsigned short)(v1 >> 16);
            }
            if (kb + 32 < K) {                      // prefetch next B tile (L2-resident weights)
                __builtin_prefetch(Bp0 + (size_t)(kb + 32 + kk) * N + nbase + 2 * c, 0, 0);
                if (DUAL) __builtin_prefetch(Bp1 + (size_t)(kb + 32 + kk) * N + nbase + 2 * c, 0, 0);
            }
        }
#if HAS_ASYNC_LDS
        __builtin_amdgcn_s_wait_asynccnt(0);
#endif
        __syncthreads();
        v16bf a0 = load_frag(&As[(wm * 32) * 40], 40);
        v16bf a1 = load_frag(&As[(wm * 32 + 16) * 40], 40);
        v16bf b0 = load_frag(&Bt[(wn * 32) * 40], 40);
        v16bf b1 = load_frag(&Bt[(wn * 32 + 16) * 40], 40);
        acc[0][0] = wmma_bf16(a0, b0, acc[0][0]);
        acc[0][1] = wmma_bf16(a0, b1, acc[0][1]);
        acc[1][0] = wmma_bf16(a1, b0, acc[1][0]);
        acc[1][1] = wmma_bf16(a1, b1, acc[1][1]);
        if (DUAL) {
            v16bf d0 = load_frag(&B1t[(wn * 32) * 40], 40);
            v16bf d1 = load_frag(&B1t[(wn * 32 + 16) * 40], 40);
            acu[0][0] = wmma_bf16(a0, d0, acu[0][0]);
            acu[0][1] = wmma_bf16(a0, d1, acu[0][1]);
            acu[1][0] = wmma_bf16(a1, d0, acu[1][0]);
            acu[1][1] = wmma_bf16(a1, d1, acu[1][1]);
        }
    }
    #pragma unroll
    for (int mi = 0; mi < 2; ++mi)
    #pragma unroll
    for (int ni = 0; ni < 2; ++ni)
    #pragma unroll
    for (int i = 0; i < 8; ++i) {
        int row = mbase + wm * 32 + mi * 16 + i + ((lane >> 4) << 3);
        int col = nbase + wn * 32 + ni * 16 + (lane & 15);
        if (row >= Mvalid) continue;
        float g = acc[mi][ni][i];
        if (MODE == 0) {
            float v = g;
            if (DUAL) { float u = acu[mi][ni][i]; v = (g / (1.0f + __expf(-g))) * u; }
            ((unsigned short*)Cout)[(size_t)z * cExpStride + (size_t)row * N + col] = f2bf(v);
        } else if (MODE == 1) {
            ((float*)Cout)[(size_t)row * N + col] = g;
        } else {
            int tok = pp[row];
            ((float*)Cout)[(size_t)tok * N + col] += wts[tok] * g;
        }
    }
}

// ---------------- Gate softmax + top-1 routing ----------------
__global__ __launch_bounds__(256) void k_gate(const unsigned short* __restrict__ xf,
        const float* __restrict__ gw, float* __restrict__ wts,
        int* __restrict__ perm, int* __restrict__ cnt) {
    __shared__ float gws[NEXP * DMODEL];
    for (int i = threadIdx.x; i < NEXP * DMODEL; i += 256) gws[i] = gw[i];
    __syncthreads();
    int t = blockIdx.x * 256 + threadIdx.x;   // exactly 8 blocks x 256 = 2048
    float lg[NEXP] = {0.f, 0.f, 0.f, 0.f};
    const unsigned short* xr = xf + (size_t)t * DMODEL;
    for (int d = 0; d < DMODEL; ++d) {
        float x = bf2f(xr[d]);
        #pragma unroll
        for (int e = 0; e < NEXP; ++e) lg[e] += x * gws[e * DMODEL + d];
    }
    float mx = lg[0];
    #pragma unroll
    for (int e = 1; e < NEXP; ++e) mx = fmaxf(mx, lg[e]);
    float sum = 0.f, p[NEXP];
    #pragma unroll
    for (int e = 0; e < NEXP; ++e) { p[e] = __expf(lg[e] - mx); sum += p[e]; }
    int best = 0; float bp = p[0];
    #pragma unroll
    for (int e = 1; e < NEXP; ++e) if (p[e] > bp) { bp = p[e]; best = e; }
    wts[t] = (bp / sum) * 1.0f;    // ROUTED_SCALE = 1.0
    int pos = atomicAdd(&cnt[best], 1);
    perm[best * TOK + pos] = t;
}

extern "C" void kernel_launch(void* const* d_in, const int* in_sizes, int n_in,
                              void* d_out, int out_size, void* d_ws, size_t ws_size,
                              hipStream_t stream) {
    (void)in_sizes; (void)n_in; (void)out_size; (void)ws_size;
    const float* dec   = (const float*)d_in[0];
    const float* wq_a  = (const float*)d_in[1];
    const float* qnw   = (const float*)d_in[2];
    const float* wq_b  = (const float*)d_in[3];
    const float* wkv_a = (const float*)d_in[4];
    const float* kvnw  = (const float*)d_in[5];
    const float* wkv_b = (const float*)d_in[6];
    const float* wo    = (const float*)d_in[7];
    const float* gw    = (const float*)d_in[8];
    const float* eg    = (const float*)d_in[9];
    const float* eu    = (const float*)d_in[10];
    const float* ed    = (const float*)d_in[11];
    const float* shg   = (const float*)d_in[12];
    const float* shu   = (const float*)d_in[13];
    const float* shd   = (const float*)d_in[14];
    float* out = (float*)d_out;

    char* w = (char*)d_ws;
    auto alloc = [&](size_t bytes) { char* p = w; w += (bytes + 255) & ~(size_t)255; return p; };
    float*          qn    = (float*)alloc((size_t)TOK * QLORA * 4);
    float*          ckvn  = (float*)alloc((size_t)TOK * KVLORA * 4);
    float*          kpe   = (float*)alloc((size_t)TOK * ROPE_D * 4);
    unsigned short* qf    = (unsigned short*)alloc((size_t)BATCH * NHEAD * SEQ * 64 * 2);
    unsigned short* kf    = (unsigned short*)alloc((size_t)BATCH * NHEAD * SEQ * 64 * 2);
    unsigned short* vt    = (unsigned short*)alloc((size_t)BATCH * NHEAD * 64 * SEQ * 2);
    unsigned short* xattn = (unsigned short*)alloc((size_t)TOK * DMODEL * 2);
    unsigned short* xf    = (unsigned short*)alloc((size_t)TOK * DMODEL * 2);
    unsigned short* wo_bf = (unsigned short*)alloc((size_t)DMODEL * DMODEL * 2);
    unsigned short* eg_bf = (unsigned short*)alloc((size_t)NEXP * DMODEL * HID * 2);
    unsigned short* eu_bf = (unsigned short*)alloc((size_t)NEXP * DMODEL * HID * 2);
    unsigned short* ed_bf = (unsigned short*)alloc((size_t)NEXP * HID * DMODEL * 2);
    unsigned short* shg_bf= (unsigned short*)alloc((size_t)DMODEL * HID * 2);
    unsigned short* shu_bf= (unsigned short*)alloc((size_t)DMODEL * HID * 2);
    unsigned short* shd_bf= (unsigned short*)alloc((size_t)HID * DMODEL * 2);
    unsigned short* hs    = (unsigned short*)alloc((size_t)TOK * HID * 2);
    unsigned short* hb    = (unsigned short*)alloc((size_t)NEXP * TOK * HID * 2);
    int*            perm  = (int*)alloc((size_t)NEXP * TOK * 4);
    int*            cnt   = (int*)alloc(256);
    float*          wts   = (float*)alloc((size_t)TOK * 4);

    auto conv = [&](const float* s, unsigned short* d, int n) {
        k_f2bf<<<(n + 255) / 256, 256, 0, stream>>>(s, d, n);
    };
    conv(wo,  wo_bf,  DMODEL * DMODEL);
    conv(eg,  eg_bf,  NEXP * DMODEL * HID);
    conv(eu,  eu_bf,  NEXP * DMODEL * HID);
    conv(ed,  ed_bf,  NEXP * HID * DMODEL);
    conv(shg, shg_bf, DMODEL * HID);
    conv(shu, shu_bf, DMODEL * HID);
    conv(shd, shd_bf, HID * DMODEL);

    k_lora<<<TOK, 256, 0, stream>>>(dec, wq_a, qnw, wkv_a, kvnw, qn, ckvn, kpe);
    k_qkv_rope<<<TOK, 256, 0, stream>>>(qn, ckvn, kpe, wq_b, wkv_b, qf, kf, vt);
    k_flash<<<dim3(SEQ / 64, BATCH * NHEAD), 128, 0, stream>>>(qf, kf, vt, xattn);

    // y = attn @ wo  -> xf (bf16, rows already (s,b)-ordered)
    k_gemm<0, false, false><<<dim3(DMODEL / 64, TOK / 64, 1), 128, 0, stream>>>(
        xattn, wo_bf, nullptr, xf, nullptr, nullptr, nullptr,
        TOK, DMODEL, DMODEL, 0, 0, 0, 0);

    (void)hipMemsetAsync(cnt, 0, NEXP * sizeof(int), stream);
    k_gate<<<TOK / 256, 256, 0, stream>>>(xf, gw, wts, perm, cnt);

    // shared: hs = silu(xf@shg) * (xf@shu);  out = hs @ shd (full write)
    k_gemm<0, true, false><<<dim3(HID / 64, TOK / 64, 1), 128, 0, stream>>>(
        xf, shg_bf, shu_bf, hs, nullptr, nullptr, nullptr,
        TOK, HID, DMODEL, 0, 0, 0, 0);
    k_gemm<1, false, false><<<dim3(DMODEL / 64, TOK / 64, 1), 128, 0, stream>>>(
        hs, shd_bf, nullptr, out, nullptr, nullptr, nullptr,
        TOK, DMODEL, HID, 0, 0, 0, 0);

    // routed experts: gathered gate/up fused, then scatter-add down-proj
    k_gemm<0, true, true><<<dim3(HID / 64, TOK / 64, NEXP), 128, 0, stream>>>(
        xf, eg_bf, eu_bf, hb, perm, cnt, nullptr,
        TOK, HID, DMODEL, 0, (long)DMODEL * HID, (long)TOK * HID, TOK);
    k_gemm<2, false, false><<<dim3(DMODEL / 64, TOK / 64, NEXP), 128, 0, stream>>>(
        hb, ed_bf, nullptr, out, perm, cnt, wts,
        TOK, DMODEL, HID, (long)TOK * HID, (long)HID * DMODEL, 0, TOK);
}
